// MedicalKnowledgeRetriever_34737695490423
// MI455X (gfx1250) — compile-verified, matrix operand order checked by reading, not run on hardware
//
#include <hip/hip_runtime.h>

#define BQ    1024
#define KDIM  768
#define NKB   100000
#define TK    3
#define HD    4096
#define KFLAT (KDIM*TK)   // 2304

typedef __attribute__((ext_vector_type(16))) __bf16 v16bf;
typedef __attribute__((ext_vector_type(8)))  __bf16 v8bf;
typedef __attribute__((ext_vector_type(8)))  float  v8f;
typedef __attribute__((ext_vector_type(4)))  unsigned v4u;
typedef __attribute__((ext_vector_type(8)))  int    v8i;
typedef __attribute__((ext_vector_type(4)))  int    v4i;

// ---------------- helpers ----------------

static __device__ __forceinline__ __bf16 f2bf(float f) {
  union { float f; unsigned u; } a; a.f = f;
  unsigned u = a.u;
  unsigned r = u + 0x7FFFu + ((u >> 16) & 1u);   // round-to-nearest-even
  unsigned short h = (unsigned short)(r >> 16);
  union { unsigned short s; __bf16 b; } o; o.s = h;
  return o.b;
}

// Load a 16-lane bf16 WMMA fragment for a row-major K-contiguous operand.
// ISA 16-bit A/B layout: lanes 0-15 hold K {k..k+7, k+16..k+23},
// lanes 16-31 hold K {k+8..k+15, k+24..k+31}.  (row = lane & 15)
static __device__ __forceinline__ v16bf load_frag(const __bf16* row, int k0, int lhi) {
  const int o1 = k0 + (lhi ? 8 : 0);
  const int o2 = k0 + (lhi ? 24 : 16);
  union { v16bf v; v8bf h[2]; } u;
  u.h[0] = *(const v8bf*)(row + o1);
  u.h[1] = *(const v8bf*)(row + o2);
  return u.v;
}

static __device__ __forceinline__ void top3_insert(float v, int i,
    float& v0, int& i0, float& v1, int& i1, float& v2, int& i2) {
  if (v > v2) {
    if (v > v0)      { v2=v1; i2=i1; v1=v0; i1=i0; v0=v; i0=i; }
    else if (v > v1) { v2=v1; i2=i1; v1=v;  i1=i; }
    else             { v2=v;  i2=i; }
  }
}

// ---------------- elementwise / layout kernels ----------------

__global__ __launch_bounds__(256) void convert_bf16_kernel(
    const float* __restrict__ in, __bf16* __restrict__ out, int n) {
  int t = blockIdx.x * 256 + threadIdx.x;
  if (t < n) out[t] = f2bf(in[t]);
}

// W[K][N] (row-major) -> Wt[N][K] bf16 (row-major)
__global__ __launch_bounds__(256) void transpose_convert_kernel(
    const float* __restrict__ W, __bf16* __restrict__ Wt, int K, int N) {
  __shared__ float tile[32][33];
  const int k0 = blockIdx.y * 32, n0 = blockIdx.x * 32;
  const int tx = threadIdx.x & 31, ty = threadIdx.x >> 5;   // 32 x 8
  for (int i = ty; i < 32; i += 8)
    if (k0 + i < K && n0 + tx < N)
      tile[i][tx] = W[(long long)(k0 + i) * N + n0 + tx];
  __syncthreads();
  for (int i = ty; i < 32; i += 8)
    if (n0 + i < N && k0 + tx < K)
      Wt[(long long)(n0 + i) * K + k0 + tx] = f2bf(tile[tx][i]);
}

// out[row] = bf16( in[row] / ||in[row]||2 ), one block per row
__global__ __launch_bounds__(256) void rownorm_bf16_kernel(
    const float* __restrict__ in, __bf16* __restrict__ out, int cols) {
  const int row = blockIdx.x, tid = threadIdx.x;
  const float* r = in + (long long)row * cols;
  float ss = 0.f;
  for (int c = tid; c < cols; c += 256) { float v = r[c]; ss += v * v; }
  __shared__ float red[256];
  red[tid] = ss; __syncthreads();
  for (int s = 128; s > 0; s >>= 1) { if (tid < s) red[tid] += red[tid + s]; __syncthreads(); }
  const float inv = rsqrtf(red[0]);
  __bf16* o = out + (long long)row * cols;
  for (int c = tid; c < cols; c += 256) o[c] = f2bf(r[c] * inv);
}

// ---------------- generic WMMA GEMM: C[M][N] = A[M][K] @ Bt[N][K]^T + bias ----------------
// grid = (N/256, M/16), block 256: 8 waves x 32 cols (2 accumulator chains/wave)
__global__ __launch_bounds__(256) void wmma_gemm_bias_kernel(
    const __bf16* __restrict__ A, const __bf16* __restrict__ Bt,
    const float* __restrict__ bias, float* __restrict__ C,
    int M, int N, int K) {
  const int tid = threadIdx.x;
  const int wave = tid >> 5, lane = tid & 31;
  const int lhi = lane >> 4, nlane = lane & 15;
  const int mBase = blockIdx.y * 16;
  const int nBase = blockIdx.x * 256 + wave * 32;

  const __bf16* arow  = A  + (long long)(mBase + nlane) * K;
  const __bf16* brow0 = Bt + (long long)(nBase + nlane) * K;
  const __bf16* brow1 = Bt + (long long)(nBase + 16 + nlane) * K;

  v8f acc0 = {}, acc1 = {};
  for (int k = 0; k < K; k += 32) {
    v16bf af  = load_frag(arow,  k, lhi);
    v16bf bf0 = load_frag(brow0, k, lhi);
    v16bf bf1 = load_frag(brow1, k, lhi);
    acc0 = __builtin_amdgcn_wmma_f32_16x16x32_bf16(false, af, false, bf0,
                                                   (short)0, acc0, false, false);
    acc1 = __builtin_amdgcn_wmma_f32_16x16x32_bf16(false, af, false, bf1,
                                                   (short)0, acc1, false, false);
  }
  const int col0 = nBase + nlane;
  const int col1 = col0 + 16;
  const float b0 = bias[col0], b1v = bias[col1];
  #pragma unroll
  for (int r = 0; r < 8; r++) {
    const int m = mBase + r + (lhi ? 8 : 0);
    C[(long long)m * N + col0] = acc0[r] + b0;
    C[(long long)m * N + col1] = acc1[r] + b1v;
  }
}

// ---------------- fused similarity + top-3 ----------------
// grid = B/16 = 64 blocks, block 256.  Each block: 16 query rows vs all NKB,
// 256-column chunks (8 waves x 32 cols, 2 accumulator chains per wave).
__global__ __launch_bounds__(256) void sim_topk_kernel(
    const __bf16* __restrict__ q,    // [BQ][KDIM] normalized
    const __bf16* __restrict__ kbn,  // [NKB][KDIM] normalized
    float* __restrict__ topv, int* __restrict__ topi,
    float* __restrict__ out_scores, int* __restrict__ out_idx) {
  __shared__ __align__(16) __bf16 qs[16 * KDIM];       // 24 KB
  __shared__ float simbuf[16 * 256];                   // 16 KB
  __shared__ float candv[16][16 * TK];
  __shared__ int   candi[16][16 * TK];

  const int tid  = threadIdx.x;
  const int wave = tid >> 5, lane = tid & 31;
  const int lhi  = lane >> 4, nlane = lane & 15;
  const int mBase = blockIdx.x * 16;

  // ---- stage 16x768 bf16 q tile into LDS with the Tensor Data Mover ----
  // D# per CDNA5 ISA: 1 descriptor, type=2, data_size=4B,
  // 1-D tile of 6144 dwords (24 KB contiguous).
  if (tid < 32) {                       // one wave issues (TDM ignores EXEC)
    const unsigned long long ga =
        (unsigned long long)(uintptr_t)(q + (long long)mBase * KDIM);
    const unsigned lds_addr = (unsigned)(uintptr_t)(void*)qs; // low 32b = LDS offset
    v4u g0;
    g0[0] = 1u;                                   // count=1, is_restore=0
    g0[1] = lds_addr;                             // lds_addr
    g0[2] = (unsigned)(ga & 0xFFFFFFFFu);         // global_addr[31:0]
    g0[3] = (unsigned)((ga >> 32) & 0x01FFFFFFu) | 0x80000000u; // addr[56:32], type=2
    v8i g1;
    g1[0] = 0x00020000;          // workgroup_mask=0, data_size=2 (4B)
    g1[1] = (int)(6144u << 16);  // tensor_dim0[15:0]=6144 in [31:16]
    g1[2] = 0x00010000;          // tensor_dim0[31:16]=0, tensor_dim1=1
    g1[3] = (int)(6144u << 16);  // tile_dim0=6144 in [31:16]
    g1[4] = 0x00000001;          // tile_dim1=1, tile_dim2=0
    g1[5] = 6144;                // tensor_dim0_stride[31:0]
    g1[6] = 0;
    g1[7] = 0;
    v4i gz4 = {};
    v8i gz8 = {};
    __builtin_amdgcn_tensor_load_to_lds(g0, g1, gz4, gz4, gz8, 0);
    __builtin_amdgcn_s_wait_tensorcnt(0);
  }
  __syncthreads();

  const int row = tid & 15;     // row this thread scans
  const int seg = tid >> 4;     // column segment (16 cols) within 256-chunk
  float bv0 = -INFINITY, bv1 = -INFINITY, bv2 = -INFINITY;
  int   bi0 = 0, bi1 = 0, bi2 = 0;

  const __bf16* arow = qs + nlane * KDIM;

  for (int chunk = 0; chunk < NKB; chunk += 256) {
    const int col0 = chunk + wave * 32 + nlane;
    const int col1 = col0 + 16;
    const int colc0 = (col0 < NKB) ? col0 : (NKB - 1);   // clamp, no branch
    const int colc1 = (col1 < NKB) ? col1 : (NKB - 1);
    const __bf16* brow0 = kbn + (long long)colc0 * KDIM;
    const __bf16* brow1 = kbn + (long long)colc1 * KDIM;

    // hint next chunk into cache
    const int pcol = (col0 + 256 < NKB) ? col0 + 256 : (NKB - 1);
    __builtin_prefetch(kbn + (long long)pcol * KDIM, 0, 1);

    v8f acc0 = {}, acc1 = {};
    #pragma unroll 4
    for (int k = 0; k < KDIM; k += 32) {
      v16bf af  = load_frag(arow,  k, lhi);
      v16bf bf0 = load_frag(brow0, k, lhi);
      v16bf bf1 = load_frag(brow1, k, lhi);
      acc0 = __builtin_amdgcn_wmma_f32_16x16x32_bf16(false, af, false, bf0,
                                                     (short)0, acc0, false, false);
      acc1 = __builtin_amdgcn_wmma_f32_16x16x32_bf16(false, af, false, bf1,
                                                     (short)0, acc1, false, false);
    }
    const int cc0 = wave * 32 + nlane;
    #pragma unroll
    for (int r = 0; r < 8; r++) {
      const int rr = r + (lhi ? 8 : 0);
      simbuf[rr * 256 + cc0]      = acc0[r];
      simbuf[rr * 256 + cc0 + 16] = acc1[r];
    }
    __syncthreads();

    #pragma unroll
    for (int j = 0; j < 16; j++) {
      const int c = seg * 16 + j;
      const int gcol = chunk + c;
      if (gcol < NKB)
        top3_insert(simbuf[row * 256 + c], gcol, bv0, bi0, bv1, bi1, bv2, bi2);
    }
    __syncthreads();
  }

  candv[row][seg * TK + 0] = bv0; candi[row][seg * TK + 0] = bi0;
  candv[row][seg * TK + 1] = bv1; candi[row][seg * TK + 1] = bi1;
  candv[row][seg * TK + 2] = bv2; candi[row][seg * TK + 2] = bi2;
  __syncthreads();

  if (seg == 0) {   // one thread per row merges 48 candidates
    float v0 = -INFINITY, v1 = -INFINITY, v2 = -INFINITY;
    int   i0 = 0, i1 = 0, i2 = 0;
    for (int i = 0; i < 16 * TK; i++)
      top3_insert(candv[row][i], candi[row][i], v0, i0, v1, i1, v2, i2);
    const int m = mBase + row;
    topv[m * TK + 0] = v0; topv[m * TK + 1] = v1; topv[m * TK + 2] = v2;
    topi[m * TK + 0] = i0; topi[m * TK + 1] = i1; topi[m * TK + 2] = i2;
    out_scores[m * TK + 0] = v0; out_scores[m * TK + 1] = v1; out_scores[m * TK + 2] = v2;
    out_idx[m * TK + 0] = i0; out_idx[m * TK + 1] = i1; out_idx[m * TK + 2] = i2;
  }
}

// ---------------- gather retrieved rows (raw kb) -> bf16 flat ----------------
__global__ __launch_bounds__(256) void gather_bf16_kernel(
    const float* __restrict__ kb, const int* __restrict__ topi,
    __bf16* __restrict__ flat) {
  const int t = blockIdx.x * 256 + threadIdx.x;
  if (t >= BQ * KFLAT) return;
  const int b = t / KFLAT;
  const int j = t - b * KFLAT;
  const int k = j / KDIM;
  const int c = j - k * KDIM;
  const int rowidx = topi[b * TK + k];
  flat[t] = f2bf(kb[(long long)rowidx * KDIM + c]);
}

// ---------------- LayerNorm (+ exact GELU) kernels, one block per row ----------------
__global__ __launch_bounds__(256) void ln_gelu_bf16_kernel(
    const float* __restrict__ x, const float* __restrict__ g,
    const float* __restrict__ be, __bf16* __restrict__ out, int N) {
  const int row = blockIdx.x, tid = threadIdx.x;
  const float* r = x + (long long)row * N;
  float s = 0.f, s2 = 0.f;
  for (int c = tid; c < N; c += 256) { float v = r[c]; s += v; s2 += v * v; }
  __shared__ float rs[256], rs2[256];
  rs[tid] = s; rs2[tid] = s2; __syncthreads();
  for (int k = 128; k > 0; k >>= 1) {
    if (tid < k) { rs[tid] += rs[tid + k]; rs2[tid] += rs2[tid + k]; }
    __syncthreads();
  }
  const float mu   = rs[0] / N;
  const float var  = rs2[0] / N - mu * mu;
  const float rstd = rsqrtf(var + 1e-5f);
  __bf16* o = out + (long long)row * N;
  for (int c = tid; c < N; c += 256) {
    float y = (r[c] - mu) * rstd * g[c] + be[c];
    float ge = 0.5f * y * (1.f + erff(y * 0.70710678118654752f));  // exact GELU
    o[c] = f2bf(ge);
  }
}

__global__ __launch_bounds__(256) void ln_f32_kernel(
    const float* __restrict__ x, const float* __restrict__ g,
    const float* __restrict__ be, float* __restrict__ out, int N) {
  const int row = blockIdx.x, tid = threadIdx.x;
  const float* r = x + (long long)row * N;
  float s = 0.f, s2 = 0.f;
  for (int c = tid; c < N; c += 256) { float v = r[c]; s += v; s2 += v * v; }
  __shared__ float rs[256], rs2[256];
  rs[tid] = s; rs2[tid] = s2; __syncthreads();
  for (int k = 128; k > 0; k >>= 1) {
    if (tid < k) { rs[tid] += rs[tid + k]; rs2[tid] += rs2[tid + k]; }
    __syncthreads();
  }
  const float mu   = rs[0] / N;
  const float var  = rs2[0] / N - mu * mu;
  const float rstd = rsqrtf(var + 1e-5f);
  float* o = out + (long long)row * N;
  for (int c = tid; c < N; c += 256)
    o[c] = (r[c] - mu) * rstd * g[c] + be[c];
}

// ---------------- launch ----------------

extern "C" void kernel_launch(void* const* d_in, const int* in_sizes, int n_in,
                              void* d_out, int out_size, void* d_ws, size_t ws_size,
                              hipStream_t stream) {
  const float* query = (const float*)d_in[0];
  const float* kb    = (const float*)d_in[1];
  const float* qW    = (const float*)d_in[2];
  const float* qb    = (const float*)d_in[3];
  const float* W1    = (const float*)d_in[4];
  const float* b1    = (const float*)d_in[5];
  const float* g1    = (const float*)d_in[6];
  const float* be1   = (const float*)d_in[7];
  const float* W2    = (const float*)d_in[8];
  const float* b2    = (const float*)d_in[9];
  const float* g2    = (const float*)d_in[10];
  const float* be2   = (const float*)d_in[11];

  // workspace carve (256B aligned)
  char* w = (char*)d_ws;
  size_t off = 0;
  auto carve = [&](size_t bytes) {
    char* p = w + off;
    off = (off + bytes + 255) & ~(size_t)255;
    return p;
  };
  __bf16* kbn_bf  = (__bf16*)carve((size_t)NKB * KDIM * 2);
  __bf16* qe_bf   = (__bf16*)carve((size_t)BQ * KDIM * 2);
  __bf16* qWt_bf  = (__bf16*)carve((size_t)KDIM * KDIM * 2);
  float*  q_f32   = (float*) carve((size_t)BQ * KDIM * 4);
  __bf16* q_bf    = (__bf16*)carve((size_t)BQ * KDIM * 2);
  __bf16* W1t_bf  = (__bf16*)carve((size_t)HD * KFLAT * 2);
  __bf16* W2t_bf  = (__bf16*)carve((size_t)HD * HD * 2);
  __bf16* flat_bf = (__bf16*)carve((size_t)BQ * KFLAT * 2);
  float*  h1_f32  = (float*) carve((size_t)BQ * HD * 4);
  __bf16* h_bf    = (__bf16*)carve((size_t)BQ * HD * 2);
  float*  h2_f32  = (float*) carve((size_t)BQ * HD * 4);
  float*  topv    = (float*) carve((size_t)BQ * TK * 4);
  int*    topi    = (int*)   carve((size_t)BQ * TK * 4);

  float* out_ctx    = (float*)d_out;                    // [BQ][HD]
  int*   out_idx    = (int*)(out_ctx + (size_t)BQ * HD);// [BQ][TK]
  float* out_scores = (float*)(out_idx + (size_t)BQ * TK);

  // 1) precision conversions
  {
    int n = BQ * KDIM;
    convert_bf16_kernel<<<(n + 255) / 256, 256, 0, stream>>>(query, qe_bf, n);
  }
  transpose_convert_kernel<<<dim3(KDIM / 32, KDIM / 32), 256, 0, stream>>>(qW, qWt_bf, KDIM, KDIM);
  transpose_convert_kernel<<<dim3(HD / 32, KFLAT / 32), 256, 0, stream>>>(W1, W1t_bf, KFLAT, HD);
  transpose_convert_kernel<<<dim3(HD / 32, HD / 32), 256, 0, stream>>>(W2, W2t_bf, HD, HD);

  // 2) kbn = normalize(kb) in bf16 (154 MB -> fits the 192 MB L2)
  rownorm_bf16_kernel<<<NKB, 256, 0, stream>>>(kb, kbn_bf, KDIM);

  // 3) q = query @ qW + qb   (WMMA), then row-normalize to bf16
  wmma_gemm_bias_kernel<<<dim3(KDIM / 256, BQ / 16), 256, 0, stream>>>(
      qe_bf, qWt_bf, qb, q_f32, BQ, KDIM, KDIM);
  rownorm_bf16_kernel<<<BQ, 256, 0, stream>>>(q_f32, q_bf, KDIM);

  // 4) fused cosine-sim GEMM + top-3 (never materializes [B, NKB])
  sim_topk_kernel<<<BQ / 16, 256, 0, stream>>>(q_bf, kbn_bf, topv, topi,
                                               out_scores, out_idx);

  // 5) gather raw kb rows -> flat bf16
  {
    int n = BQ * KFLAT;
    gather_bf16_kernel<<<(n + 255) / 256, 256, 0, stream>>>(kb, topi, flat_bf);
  }

  // 6) h = gelu(LN(flat @ W1 + b1))
  wmma_gemm_bias_kernel<<<dim3(HD / 256, BQ / 16), 256, 0, stream>>>(
      flat_bf, W1t_bf, b1, h1_f32, BQ, HD, KFLAT);
  ln_gelu_bf16_kernel<<<BQ, 256, 0, stream>>>(h1_f32, g1, be1, h_bf, HD);

  // 7) context = LN(h @ W2 + b2)
  wmma_gemm_bias_kernel<<<dim3(HD / 256, BQ / 16), 256, 0, stream>>>(
      h_bf, W2t_bf, b2, h2_f32, BQ, HD, HD);
  ln_f32_kernel<<<BQ, 256, 0, stream>>>(h2_f32, g2, be2, out_ctx, HD);

  (void)in_sizes; (void)n_in; (void)out_size; (void)ws_size;
}
